// gnn_rnn_diagnoser_62852551410035
// MI455X (gfx1250) — compile-verified, matrix operand order checked by reading, not run on hardware
//
#include <hip/hip_runtime.h>
#include <stdint.h>
#include <math.h>

typedef __bf16 bf16_t;
typedef __attribute__((ext_vector_type(16))) __bf16 v16bf;
typedef __attribute__((ext_vector_type(8)))  float  v8f;

#define NODE 5
#define BT   128        // batch rows per workgroup
#define TSTEPS 128
#define SSP  328        // padded row stride (elems) for state tiles (328*2B = 164 dw % 64 = 36)
#define SHP  264        // padded row stride (elems) for hidden tiles (264*2B = 132 dw % 64 = 4)

struct __align__(16) U128 { unsigned x, y, z, w; };
union FragAB { v16bf v; U128 q[2]; };

__device__ __forceinline__ bf16_t f2bf(float f) { return (bf16_t)f; }

__device__ __forceinline__ v8f wmma_bf16(const FragAB& a, const FragAB& b, v8f c) {
  return __builtin_amdgcn_wmma_f32_16x16x32_bf16(false, a.v, false, b.v,
                                                 (short)0, c, false, false);
}

__device__ __forceinline__ void load_frag(FragAB& f, const bf16_t* base) {
  f.q[0] = *(const U128*)(base);
  f.q[1] = *(const U128*)(base + 16);
}

// ---------------------------------------------------------------------------
// Prologue: transpose + convert all weights to bf16 [out][k] layout.
// ---------------------------------------------------------------------------
__global__ __launch_bounds__(256) void transpose_weights(
    const float* __restrict__ W1, const float* __restrict__ W2,
    const float* __restrict__ W3, const float* __restrict__ Wo1,
    bf16_t* __restrict__ W1t, bf16_t* __restrict__ W2t,
    bf16_t* __restrict__ W3t, bf16_t* __restrict__ Wo1t) {
  const int N1 = NODE * 256 * 128;
  const int N2 = NODE * 256 * 256;
  const int N3 = NODE * 64 * 256;
  const int N4 = 256 * 320;
  int gid = blockIdx.x * 256 + threadIdx.x;
  if (gid < N1) {
    int n = gid / (256 * 128), r = gid % (256 * 128), o = r / 128, k = r % 128;
    W1t[gid] = f2bf(W1[((size_t)n * 130 + 2 + k) * 256 + o]);
  } else if (gid < N1 + N2) {
    int g = gid - N1;
    int n = g / 65536, r = g % 65536, o = r / 256, k = r % 256;
    W2t[g] = f2bf(W2[((size_t)n * 256 + k) * 256 + o]);
  } else if (gid < N1 + N2 + N3) {
    int g = gid - N1 - N2;
    int n = g / 16384, r = g % 16384, o = r / 256, k = r % 256;
    W3t[g] = f2bf(W3[((size_t)n * 256 + k) * 64 + o]);
  } else if (gid < N1 + N2 + N3 + N4) {
    int g = gid - N1 - N2 - N3;
    int o = g / 320, k = g % 320;
    Wo1t[g] = f2bf(Wo1[(size_t)(5 + k) * 256 + o]);
  }
}

// ---------------------------------------------------------------------------
// Main recurrent scan. One workgroup = 128 batch rows, persistent over all T.
// Each wave owns output columns (B-fragments loaded once per layer pass) and
// computes two M-tiles simultaneously (two independent WMMA chains).
// LDS rows padded by 16B so half-wave fragment loads cover all 64 banks.
// ---------------------------------------------------------------------------
__global__ __launch_bounds__(256) void rnn_scan_kernel(
    const float* __restrict__ x,        // (B, 5, T)
    const float* __restrict__ W1,       // (5, 130, 256) f32 (rows 0,1 used)
    const float* __restrict__ b1, const float* __restrict__ b2,
    const float* __restrict__ b3,
    const bf16_t* __restrict__ W1t,     // (5, 256, 128)
    const bf16_t* __restrict__ W2t,     // (5, 256, 256)
    const bf16_t* __restrict__ W3t,     // (5,  64, 256)
    bf16_t* __restrict__ state_out,     // (B, 320)
    int B, int T) {
  __shared__ __align__(16) bf16_t s_state[2][BT * SSP];  // [row][node*64+f]
  __shared__ __align__(16) bf16_t s_h[2][BT * SHP];
  __shared__ float s_x[NODE][BT];

  const int tid  = threadIdx.x;
  const int wave = tid >> 5;
  const int lane = tid & 31;
  const int lm   = lane & 15;   // M (A/C) or N (B/C) index within 16x16 tile
  const int lh   = lane >> 4;   // lane-half selector
  const int kb   = lh * 8;      // per-lane K sub-chunk base
  const int row0 = blockIdx.x * BT;

  const int PAR[NODE][2] = {{3, 4}, {0, 4}, {0, 1}, {1, 2}, {2, 3}};

  // zero initial state (buffer 0, incl. padding)
  for (int i = tid; i < BT * SSP / 8; i += 256)
    ((U128*)s_state[0])[i] = U128{0u, 0u, 0u, 0u};
  __syncthreads();

  for (int t = 0; t < T; ++t) {
    const int cur = t & 1, nxt = cur ^ 1;

    // stage x_t for all 5 nodes, 128 rows
    for (int i = tid; i < NODE * BT; i += 256) {
      int n = i / BT, r = i % BT;
      s_x[n][r] = x[((size_t)(row0 + r) * NODE + n) * T + t];
    }
    __syncthreads();

    for (int n = 0; n < NODE; ++n) {
      const int p0 = PAR[n][0], p1 = PAR[n][1];

      // ---- layer 1: (BT x 128) @ (128 x 256) + rank-2 x-term, relu ----
      for (int c = 0; c < 2; ++c) {
        const int nc = wave * 32 + c * 16 + lm;
        FragAB Bf[4];
        const bf16_t* wr = &W1t[((size_t)n * 256 + nc) * 128];
#pragma unroll
        for (int j = 0; j < 4; ++j) load_frag(Bf[j], wr + j * 32 + kb);
        const float bv = b1[n * 256 + nc];
        const float w0 = W1[((size_t)n * 130 + 0) * 256 + nc];
        const float w1 = W1[((size_t)n * 130 + 1) * 256 + nc];
        for (int mp = 0; mp < 4; ++mp) {
          const int ar0 = (mp * 2) * 16 + lm, ar1 = ar0 + 16;
          v8f acc0, acc1;
#pragma unroll
          for (int r = 0; r < 8; ++r) { acc0[r] = bv; acc1[r] = bv; }
#pragma unroll
          for (int j = 0; j < 4; ++j) {
            const int off = ((j < 2) ? p0 : p1) * 64 + (j & 1) * 32 + kb;
            FragAB a0, a1;
            load_frag(a0, &s_state[cur][ar0 * SSP + off]);
            load_frag(a1, &s_state[cur][ar1 * SSP + off]);
            acc0 = wmma_bf16(a0, Bf[j], acc0);
            acc1 = wmma_bf16(a1, Bf[j], acc1);
          }
#pragma unroll
          for (int r = 0; r < 8; ++r) {
            const int rw0 = (mp * 2) * 16 + 8 * lh + r;
            const int rw1 = rw0 + 16;
            float v0 = acc0[r] + s_x[p0][rw0] * w0 + s_x[p1][rw0] * w1;
            float v1 = acc1[r] + s_x[p0][rw1] * w0 + s_x[p1][rw1] * w1;
            s_h[0][rw0 * SHP + nc] = f2bf(fmaxf(v0, 0.f));
            s_h[0][rw1 * SHP + nc] = f2bf(fmaxf(v1, 0.f));
          }
        }
      }
      __syncthreads();

      // ---- layer 2: (BT x 256) @ (256 x 256), relu ----
      for (int c = 0; c < 2; ++c) {
        const int nc = wave * 32 + c * 16 + lm;
        FragAB Bf[8];
        const bf16_t* wr = &W2t[((size_t)n * 256 + nc) * 256];
#pragma unroll
        for (int j = 0; j < 8; ++j) load_frag(Bf[j], wr + j * 32 + kb);
        const float bv = b2[n * 256 + nc];
        for (int mp = 0; mp < 4; ++mp) {
          const int ar0 = (mp * 2) * 16 + lm, ar1 = ar0 + 16;
          v8f acc0, acc1;
#pragma unroll
          for (int r = 0; r < 8; ++r) { acc0[r] = bv; acc1[r] = bv; }
#pragma unroll
          for (int j = 0; j < 8; ++j) {
            FragAB a0, a1;
            load_frag(a0, &s_h[0][ar0 * SHP + j * 32 + kb]);
            load_frag(a1, &s_h[0][ar1 * SHP + j * 32 + kb]);
            acc0 = wmma_bf16(a0, Bf[j], acc0);
            acc1 = wmma_bf16(a1, Bf[j], acc1);
          }
#pragma unroll
          for (int r = 0; r < 8; ++r) {
            const int rw0 = (mp * 2) * 16 + 8 * lh + r;
            const int rw1 = rw0 + 16;
            s_h[1][rw0 * SHP + nc] = f2bf(fmaxf(acc0[r], 0.f));
            s_h[1][rw1 * SHP + nc] = f2bf(fmaxf(acc1[r], 0.f));
          }
        }
      }
      __syncthreads();

      // ---- layer 3: (BT x 256) @ (256 x 64), tanh -> next state ----
      {
        const int nc = (wave & 3) * 16 + lm;     // 4 N-tiles
        const int mb = (wave >> 2) * 4;          // 2 M-halves
        FragAB Bf[8];
        const bf16_t* wr = &W3t[((size_t)n * 64 + nc) * 256];
#pragma unroll
        for (int j = 0; j < 8; ++j) load_frag(Bf[j], wr + j * 32 + kb);
        const float bv = b3[n * 64 + nc];
        for (int mp = 0; mp < 2; ++mp) {
          const int m0 = mb + mp * 2;
          const int ar0 = m0 * 16 + lm, ar1 = ar0 + 16;
          v8f acc0, acc1;
#pragma unroll
          for (int r = 0; r < 8; ++r) { acc0[r] = bv; acc1[r] = bv; }
#pragma unroll
          for (int j = 0; j < 8; ++j) {
            FragAB a0, a1;
            load_frag(a0, &s_h[1][ar0 * SHP + j * 32 + kb]);
            load_frag(a1, &s_h[1][ar1 * SHP + j * 32 + kb]);
            acc0 = wmma_bf16(a0, Bf[j], acc0);
            acc1 = wmma_bf16(a1, Bf[j], acc1);
          }
#pragma unroll
          for (int r = 0; r < 8; ++r) {
            const int rw0 = m0 * 16 + 8 * lh + r;
            const int rw1 = rw0 + 16;
            s_state[nxt][rw0 * SSP + n * 64 + nc] = f2bf(tanhf(acc0[r]));
            s_state[nxt][rw1 * SSP + n * 64 + nc] = f2bf(tanhf(acc1[r]));
          }
        }
      }
      __syncthreads();
    }  // node
  }    // t

  // write final state (buffer T&1) to global, skipping row padding
  const int fin = T & 1;
  U128* dst = (U128*)&state_out[(size_t)row0 * 320];
  const U128* src = (const U128*)s_state[fin];
  for (int i = tid; i < BT * 40; i += 256) {   // 320 elems = 40 U128 per row
    int r = i / 40, q = i % 40;
    dst[r * 40 + q] = src[r * (SSP / 8) + q];  // SSP/8 = 41 U128 per padded row
  }
}

// ---------------------------------------------------------------------------
// Head GEMM: h = relu([x_last(5) | state(320)] @ Wo1 + bo1), plus BN partials.
// ---------------------------------------------------------------------------
__global__ __launch_bounds__(256) void zero_stats(float* g_sum, float* g_sumsq) {
  int i = blockIdx.x * 256 + threadIdx.x;
  if (i < 256) g_sum[i] = 0.f;
  else if (i < 512) g_sumsq[i - 256] = 0.f;
}

__global__ __launch_bounds__(256) void head_gemm_kernel(
    const float* __restrict__ x, const float* __restrict__ Wo1,
    const float* __restrict__ bo1, const bf16_t* __restrict__ Wo1t,
    const bf16_t* __restrict__ state, float* __restrict__ h_out,
    float* __restrict__ g_sum, float* __restrict__ g_sumsq, int B, int T) {
  __shared__ __align__(16) bf16_t s_a[BT * SSP];
  __shared__ float s_xl[NODE][BT];
  __shared__ float s_sum[256], s_sq[256];

  const int tid = threadIdx.x;
  const int wave = tid >> 5, lane = tid & 31;
  const int lm = lane & 15, lh = lane >> 4, kb = lh * 8;
  const int row0 = blockIdx.x * BT;

  if (tid < 256) { s_sum[tid] = 0.f; s_sq[tid] = 0.f; }
  // stage state tile (into padded rows) + x_last into LDS
  {
    U128* dst = (U128*)s_a;
    const U128* src = (const U128*)&state[(size_t)row0 * 320];
    for (int i = tid; i < BT * 40; i += 256) {
      int r = i / 40, q = i % 40;
      dst[r * (SSP / 8) + q] = src[r * 40 + q];
    }
    for (int i = tid; i < NODE * BT; i += 256) {
      int n = i / BT, r = i % BT;
      s_xl[n][r] = x[((size_t)(row0 + r) * NODE + n) * T + (T - 1)];
    }
  }
  __syncthreads();

  for (int c = 0; c < 2; ++c) {
    const int nc = (wave * 2 + c) * 16 + lm;
    FragAB Bf[10];
    const bf16_t* wr = &Wo1t[(size_t)nc * 320];
#pragma unroll
    for (int j = 0; j < 10; ++j) load_frag(Bf[j], wr + j * 32 + kb);
    const float bv = bo1[nc];
    float wx[NODE];
#pragma unroll
    for (int p = 0; p < NODE; ++p) wx[p] = Wo1[(size_t)p * 256 + nc];
    for (int mp = 0; mp < 4; ++mp) {
      const int ar0 = (mp * 2) * 16 + lm, ar1 = ar0 + 16;
      v8f acc0, acc1;
#pragma unroll
      for (int r = 0; r < 8; ++r) { acc0[r] = bv; acc1[r] = bv; }
#pragma unroll
      for (int j = 0; j < 10; ++j) {
        FragAB a0, a1;
        load_frag(a0, &s_a[ar0 * SSP + j * 32 + kb]);
        load_frag(a1, &s_a[ar1 * SSP + j * 32 + kb]);
        acc0 = wmma_bf16(a0, Bf[j], acc0);
        acc1 = wmma_bf16(a1, Bf[j], acc1);
      }
      float lsum = 0.f, lsq = 0.f;
#pragma unroll
      for (int r = 0; r < 8; ++r) {
        const int rw0 = (mp * 2) * 16 + 8 * lh + r;
        const int rw1 = rw0 + 16;
        float v0 = acc0[r], v1 = acc1[r];
#pragma unroll
        for (int p = 0; p < NODE; ++p) {
          v0 += s_xl[p][rw0] * wx[p];
          v1 += s_xl[p][rw1] * wx[p];
        }
        v0 = fmaxf(v0, 0.f);
        v1 = fmaxf(v1, 0.f);
        h_out[(size_t)(row0 + rw0) * 256 + nc] = v0;
        h_out[(size_t)(row0 + rw1) * 256 + nc] = v1;
        lsum += v0 + v1;
        lsq += v0 * v0 + v1 * v1;
      }
      atomicAdd(&s_sum[nc], lsum);
      atomicAdd(&s_sq[nc], lsq);
    }
  }
  __syncthreads();
  if (tid < 256) {
    atomicAdd(&g_sum[tid], s_sum[tid]);
    atomicAdd(&g_sumsq[tid], s_sq[tid]);
  }
}

__global__ __launch_bounds__(256) void bn_finalize(
    const float* g_sum, const float* g_sumsq, float* mu, float* invstd, int B) {
  int i = threadIdx.x;
  if (i < 256) {
    float m = g_sum[i] / (float)B;
    float v = g_sumsq[i] / (float)B - m * m;
    mu[i] = m;
    invstd[i] = rsqrtf(v + 1e-5f);
  }
}

// ---------------------------------------------------------------------------
// Per-row: batchnorm affine, (256 @ 7), softmax.
// ---------------------------------------------------------------------------
__global__ __launch_bounds__(256) void out_kernel(
    const float* __restrict__ h, const float* __restrict__ mu,
    const float* __restrict__ invstd, const float* __restrict__ gamma,
    const float* __restrict__ beta, const float* __restrict__ Wo2,
    const float* __restrict__ bo2, float* __restrict__ out, int B) {
  __shared__ float sW[256 * 7], sb[7];
  __shared__ float sscale[256], sshift[256];
  const int tid = threadIdx.x;
  for (int i = tid; i < 256 * 7; i += 256) sW[i] = Wo2[i];
  if (tid < 7) sb[tid] = bo2[tid];
  if (tid < 256) {
    float sc = invstd[tid] * gamma[tid];
    sscale[tid] = sc;
    sshift[tid] = beta[tid] - mu[tid] * sc;
  }
  __syncthreads();

  const int row = blockIdx.x * 256 + tid;
  if (row >= B) return;
  float logit[7];
#pragma unroll
  for (int j = 0; j < 7; ++j) logit[j] = sb[j];
  const float* hr = &h[(size_t)row * 256];
  for (int f = 0; f < 256; ++f) {
    float hn = hr[f] * sscale[f] + sshift[f];
#pragma unroll
    for (int j = 0; j < 7; ++j) logit[j] += hn * sW[f * 7 + j];
  }
  float mx = logit[0];
#pragma unroll
  for (int j = 1; j < 7; ++j) mx = fmaxf(mx, logit[j]);
  float s = 0.f, e[7];
#pragma unroll
  for (int j = 0; j < 7; ++j) { e[j] = __expf(logit[j] - mx); s += e[j]; }
  float inv = 1.f / s;
#pragma unroll
  for (int j = 0; j < 7; ++j) out[(size_t)row * 7 + j] = e[j] * inv;
}

// ---------------------------------------------------------------------------
extern "C" void kernel_launch(void* const* d_in, const int* in_sizes, int n_in,
                              void* d_out, int out_size, void* d_ws, size_t ws_size,
                              hipStream_t stream) {
  const float* x     = (const float*)d_in[0];
  const float* W1    = (const float*)d_in[1];
  const float* b1    = (const float*)d_in[2];
  const float* W2    = (const float*)d_in[3];
  const float* b2    = (const float*)d_in[4];
  const float* W3    = (const float*)d_in[5];
  const float* b3    = (const float*)d_in[6];
  const float* Wo1   = (const float*)d_in[7];
  const float* bo1   = (const float*)d_in[8];
  const float* gamma = (const float*)d_in[9];
  const float* beta  = (const float*)d_in[10];
  const float* Wo2   = (const float*)d_in[11];
  const float* bo2   = (const float*)d_in[12];
  float* out = (float*)d_out;

  const int T = TSTEPS;
  const int B = in_sizes[0] / (NODE * T);

  char* ws = (char*)d_ws;
  size_t off = 0;
  auto take = [&](size_t bytes) -> char* {
    char* p = ws + off;
    off = (off + bytes + 255) & ~(size_t)255;
    return p;
  };
  bf16_t* W1t    = (bf16_t*)take((size_t)NODE * 256 * 128 * 2);
  bf16_t* W2t    = (bf16_t*)take((size_t)NODE * 256 * 256 * 2);
  bf16_t* W3t    = (bf16_t*)take((size_t)NODE * 64 * 256 * 2);
  bf16_t* Wo1t   = (bf16_t*)take((size_t)256 * 320 * 2);
  bf16_t* stateg = (bf16_t*)take((size_t)B * 320 * 2);
  float*  h_out  = (float*)take((size_t)B * 256 * 4);
  float*  g_sum  = (float*)take(256 * 4);
  float*  g_sq   = (float*)take(256 * 4);
  float*  mu     = (float*)take(256 * 4);
  float*  inv    = (float*)take(256 * 4);

  const int totT = NODE * 256 * 128 + NODE * 256 * 256 + NODE * 64 * 256 + 256 * 320;
  transpose_weights<<<(totT + 255) / 256, 256, 0, stream>>>(W1, W2, W3, Wo1,
                                                            W1t, W2t, W3t, Wo1t);
  rnn_scan_kernel<<<B / BT, 256, 0, stream>>>(x, W1, b1, b2, b3, W1t, W2t, W3t,
                                              stateg, B, T);
  zero_stats<<<2, 256, 0, stream>>>(g_sum, g_sq);
  head_gemm_kernel<<<B / BT, 256, 0, stream>>>(x, Wo1, bo1, Wo1t, stateg, h_out,
                                               g_sum, g_sq, B, T);
  bn_finalize<<<1, 256, 0, stream>>>(g_sum, g_sq, mu, inv, B);
  out_kernel<<<(B + 255) / 256, 256, 0, stream>>>(h_out, mu, inv, gamma, beta,
                                                  Wo2, bo2, out, B);
}
